// LineProposalNetwork_1932735283818
// MI455X (gfx1250) — matrix operand order repaired and checked by reading, not compiled, not run on hardware
//
#include <hip/hip_runtime.h>

// ---------------- problem constants ----------------
#define B_    4
#define C_    256
#define H_    160
#define W_    160
#define HW    25600        // H*W
#define NPIX  102400       // B*H*W
#define PW_   162          // padded width (halo 1)
#define PHW_  26244        // 162*162
#define HID_  64
#define NH_   6
#define MAXO_ 4
#define OC1   384          // NH_*HID_
#define OC2   24           // NH_*MAXO_
#define K1C   72           // (C_*9)/32 k-chunks for conv1
#define K2C   18           // (HID_*9)/32 k-chunks for conv2
#define JK    300
#define LK    5000
#define BIGK  90000        // JK*JK
#define EPSV  1e-5f

// d_out float offsets (elements)
#define O_LMAP 0
#define O_JMAP 102400
#define O_JOFF 204800
#define O_CMAP 409600
#define O_COFF 512000
#define O_LVEC 716800
#define O_LOI  1126400
#define O_MASK 1206400

// ---------------- types ----------------
typedef __bf16 bf16_t;
typedef bf16_t v16bf __attribute__((ext_vector_type(16)));
typedef float  v8f   __attribute__((ext_vector_type(8)));
typedef unsigned int u32x4 __attribute__((ext_vector_type(4)));
typedef int          i32x4 __attribute__((ext_vector_type(4)));
typedef int          i32x8 __attribute__((ext_vector_type(8)));

union FragU {
  v16bf v;
  unsigned short s[16];
  uint4 q[2];
};

__device__ __forceinline__ unsigned short f2b(float x) {
  unsigned int u = __float_as_uint(x);
  unsigned int r = (u + 0x7fffu + ((u >> 16) & 1u)) >> 16;   // RNE bf16
  return (unsigned short)r;
}
__device__ __forceinline__ float sigm(float v) { return 1.0f / (1.0f + __expf(-v)); }

#define WMMA_BF16(A, Bv, Cacc) \
  __builtin_amdgcn_wmma_f32_16x16x32_bf16(false, (A), false, (Bv), (short)0, (Cacc), false, false)

// ---------------- weight / activation packing ----------------

// fold BN scale into w1, pack into WMMA A-fragment order:
// pw1[((mt*K1C + kc)*32 + lane)*16 + slot]
__global__ void w1_pack_kernel(const float* __restrict__ w1,
                               const float* __restrict__ gamma,
                               const float* __restrict__ var,
                               unsigned short* __restrict__ pw1) {
  int t = blockIdx.x * blockDim.x + threadIdx.x;
  if (t >= 24 * K1C * 512) return;
  int slot = t & 15;
  int lane = (t >> 4) & 31;
  int grp  = t >> 9;              // mt*K1C + kc
  int kc   = grp % K1C;
  int mt   = grp / K1C;
  int m = lane & 15;
  int K = (lane >> 4) * 16 + slot;
  int tap = kc >> 3;                       // 9 taps, 8 chunks of 32ch each
  int c   = (kc & 7) * 32 + K;
  int oc  = mt * 16 + m;
  float s = gamma[oc] * rsqrtf(var[oc] + EPSV);
  float v = w1[((size_t)oc * C_ + c) * 9 + tap] * s;
  pw1[t] = f2b(v);
}

__global__ void bias_fold_kernel(const float* __restrict__ b1,
                                 const float* __restrict__ gamma,
                                 const float* __restrict__ beta,
                                 const float* __restrict__ mean,
                                 const float* __restrict__ var,
                                 float* __restrict__ bias1) {
  int i = blockIdx.x * blockDim.x + threadIdx.x;
  if (i >= OC1) return;
  float s = gamma[i] * rsqrtf(var[i] + EPSV);
  bias1[i] = (b1[i] - mean[i]) * s + beta[i];
}

// pack w2 (zero-padded OC 4->16) into WMMA B-fragment order:
// pw2[((g*K2C + kc)*32 + lane)*16 + slot]
__global__ void w2_pack_kernel(const float* __restrict__ w2,
                               unsigned short* __restrict__ pw2) {
  int t = blockIdx.x * blockDim.x + threadIdx.x;
  if (t >= NH_ * K2C * 512) return;
  int slot = t & 15;
  int lane = (t >> 4) & 31;
  int grp  = t >> 9;
  int kc   = grp % K2C;
  int g    = grp / K2C;
  int n = lane & 15;
  int K = (lane >> 4) * 8 + slot + ((slot < 8) ? 0 : 8);
  int kg  = kc * 32 + K;                    // 0..575
  int tap = kg / HID_;
  int cl  = kg % HID_;
  float v = 0.0f;
  if (n < MAXO_) v = w2[(((size_t)(g * MAXO_ + n)) * HID_ + cl) * 9 + tap];
  pw2[t] = f2b(v);
}

// features NCHW f32 -> zero-padded (162x162) NHWC bf16
__global__ void feat_pack_kernel(const float* __restrict__ f,
                                 unsigned short* __restrict__ out) {
  int t = blockIdx.x * blockDim.x + threadIdx.x;
  if (t >= B_ * PHW_ * C_) return;
  int c = t % C_;
  int pp = t / C_;
  int b = pp / PHW_, pr = pp % PHW_;
  int py = pr / PW_, px = pr % PW_;
  float v = 0.0f;
  if (py >= 1 && py <= H_ && px >= 1 && px <= W_)
    v = f[((size_t)b * C_ + c) * HW + (py - 1) * W_ + (px - 1)];
  out[t] = f2b(v);
}

// zero the padded activation buffer (border must be 0 every call)
__global__ void act_zero_kernel(uint4* __restrict__ actP4) {
  size_t t = (size_t)blockIdx.x * blockDim.x + threadIdx.x;
  if (t >= (size_t)B_ * PHW_ * OC1 / 8) return;
  actP4[t] = make_uint4(0, 0, 0, 0);
}

// ---------------- conv1: implicit GEMM, WMMA bf16 ----------------
// M=OC1, N=NPIX, K=2304. Wave: 32oc x 64pix (8 accumulators, B reused 2x).
// Block: 8 waves = 4 M-pairs x 2 N-halves -> 128oc x 128pix. Grid (800, 3).
__global__ void __launch_bounds__(256)
conv1_wmma_kernel(const unsigned short* __restrict__ featP,  // padded NHWC bf16
                  const unsigned short* __restrict__ pw1,    // fragment order
                  const float* __restrict__ bias1,
                  unsigned short* __restrict__ actP)         // padded NHWC bf16
{
  const int lane = threadIdx.x & 31;
  const int wave = threadIdx.x >> 5;
  const int hi = lane >> 4;
  const int ln = lane & 15;
  const int mt0 = blockIdx.y * 8 + (wave & 3) * 2;        // mt0, mt0+1
  const int pixBase = blockIdx.x * 128 + (wave >> 2) * 64;

  size_t ppos[4];
  const unsigned short* bp[4];
#pragma unroll
  for (int s = 0; s < 4; s++) {
    int pixel = pixBase + s * 16 + ln;
    int b = pixel / HW, rem = pixel % HW;
    int y = rem / W_, x = rem % W_;
    ppos[s] = (size_t)b * PHW_ + (size_t)(y + 1) * PW_ + (x + 1);
    bp[s] = featP + ppos[s] * C_ + hi * 16;
  }

  v8f acc[8] = {};
  const unsigned short* wpa = pw1 + ((size_t)mt0 * K1C) * 512 + lane * 16;
  const unsigned short* wpb = wpa + (size_t)K1C * 512;

  for (int tap = 0; tap < 9; tap++) {
    const int dy = tap / 3 - 1, dxv = tap % 3 - 1;
    const int doffBase = (dy * PW_ + dxv) * C_;   // halo offset, always valid (padded)
    for (int cc = 0; cc < 8; cc++) {
      const int kcIdx = tap * 8 + cc;
      v16bf a0 = *reinterpret_cast<const v16bf*>(wpa + (size_t)kcIdx * 512);
      v16bf a1 = *reinterpret_cast<const v16bf*>(wpb + (size_t)kcIdx * 512);
      if (kcIdx + 1 < K1C)
        __builtin_prefetch(wpa + (size_t)(kcIdx + 1) * 512, 0, 1);
      const int doff = doffBase + cc * 32;
      FragU b0, b1, b2, b3;
      b0.q[0] = *reinterpret_cast<const uint4*>(bp[0] + doff);
      b0.q[1] = *reinterpret_cast<const uint4*>(bp[0] + doff + 8);
      b1.q[0] = *reinterpret_cast<const uint4*>(bp[1] + doff);
      b1.q[1] = *reinterpret_cast<const uint4*>(bp[1] + doff + 8);
      b2.q[0] = *reinterpret_cast<const uint4*>(bp[2] + doff);
      b2.q[1] = *reinterpret_cast<const uint4*>(bp[2] + doff + 8);
      b3.q[0] = *reinterpret_cast<const uint4*>(bp[3] + doff);
      b3.q[1] = *reinterpret_cast<const uint4*>(bp[3] + doff + 8);
      acc[0] = WMMA_BF16(a0, b0.v, acc[0]);
      acc[1] = WMMA_BF16(a0, b1.v, acc[1]);
      acc[2] = WMMA_BF16(a0, b2.v, acc[2]);
      acc[3] = WMMA_BF16(a0, b3.v, acc[3]);
      acc[4] = WMMA_BF16(a1, b0.v, acc[4]);
      acc[5] = WMMA_BF16(a1, b1.v, acc[5]);
      acc[6] = WMMA_BF16(a1, b2.v, acc[6]);
      acc[7] = WMMA_BF16(a1, b3.v, acc[7]);
    }
  }

  // epilogue: bias + ReLU + bf16 store (8 consecutive OCs per lane)
#pragma unroll
  for (int mi = 0; mi < 2; mi++) {
    const int ocBase = (mt0 + mi) * 16 + hi * 8;
    float bb[8];
#pragma unroll
    for (int r = 0; r < 8; r++) bb[r] = bias1[ocBase + r];
#pragma unroll
    for (int s = 0; s < 4; s++) {
      v8f a = acc[mi * 4 + s];
      unsigned short outb[8];
#pragma unroll
      for (int r = 0; r < 8; r++) outb[r] = f2b(fmaxf(a[r] + bb[r], 0.0f));
      unsigned short* dst = actP + ppos[s] * OC1 + ocBase;
      *reinterpret_cast<uint4*>(dst) = *reinterpret_cast<const uint4*>(outb);
    }
  }
}

// ---------------- conv2: grouped implicit GEMM, TDM weight staging ----------------
// Per block: one group, 128 pixels (8 waves x 16). A=padded activations
// (16pix x 32K), B=packed group weights staged in LDS via tensor DMA.
__global__ void __launch_bounds__(256)
conv2_wmma_kernel(const unsigned short* __restrict__ actP,
                  const unsigned short* __restrict__ pw2,
                  const float* __restrict__ b2,
                  float* __restrict__ y2)   // (B,24,H,W) f32
{
  __shared__ __attribute__((aligned(16))) unsigned short wsm[K2C * 512]; // 18 KB
  const int lane = threadIdx.x & 31;
  const int wave = threadIdx.x >> 5;
  const int g = blockIdx.y;
  const int pixBase = blockIdx.x * 128 + wave * 16;

  if (wave == 0) {
    // Tensor Data Mover: 1D tile of 9216 bf16 elements -> LDS offset 0
    unsigned long long ga = (unsigned long long)(pw2 + (size_t)g * K2C * 512);
    u32x4 g0 = {};
    g0.x = 1u;                                         // count=1, user mode
    g0.y = 0u;                                         // lds_addr
    g0.z = (unsigned)(ga & 0xffffffffu);               // global_addr[31:0]
    g0.w = (unsigned)((ga >> 32) & 0x01ffffffu) | (2u << 30);  // addr[56:32] | type=2
    i32x8 g1 = {};
    g1[0] = 0x10000;                                   // data_size = 2B
    g1[1] = (int)(9216u << 16);                        // tensor_dim0 = 9216
    g1[2] = 0;
    g1[3] = (int)(9216u << 16);                        // tile_dim0 = 9216
    g1[4] = 1;                                         // tile_dim1 = 1
    g1[5] = 9216;                                      // tensor_dim0_stride
    g1[6] = 0; g1[7] = 0;
    i32x4 z4 = {};
#if defined(__clang_major__) && (__clang_major__ >= 23)
    i32x8 z8 = {};
    __builtin_amdgcn_tensor_load_to_lds(g0, g1, z4, z4, z8, 0);
#else
    __builtin_amdgcn_tensor_load_to_lds(g0, g1, z4, z4, 0);
#endif
    __builtin_amdgcn_s_wait_tensorcnt(0);
  }
  __syncthreads();

  const int hi = lane >> 4;
  const int ln = lane & 15;
  const int pixel = pixBase + ln;               // A row (M)
  const int b  = pixel / HW;
  const int rm = pixel % HW;
  const int py_ = rm / W_, px_ = rm % W_;
  const int K0 = hi * 8;                        // A-frag K run start
  const unsigned short* ap =
      actP + ((size_t)b * PHW_ + (size_t)(py_ + 1) * PW_ + (px_ + 1)) * OC1 + g * HID_ + K0;

  v8f acc = {};
  for (int tap = 0; tap < 9; tap++) {
    const int dy = tap / 3 - 1, dxv = tap % 3 - 1;
    const int doffBase = (dy * PW_ + dxv) * OC1;
#pragma unroll
    for (int half = 0; half < 2; half++) {
      const int kc = tap * 2 + half;
      FragU bfr;                                  // B from LDS (fragment order)
      const unsigned short* lp = wsm + kc * 512 + lane * 16;
      bfr.q[0] = *reinterpret_cast<const uint4*>(lp);
      bfr.q[1] = *reinterpret_cast<const uint4*>(lp + 8);
      FragU af;                                   // A from padded activations
      const unsigned short* src = ap + doffBase + half * 32;
      af.q[0] = *reinterpret_cast<const uint4*>(src);        // K0..K0+7
      af.q[1] = *reinterpret_cast<const uint4*>(src + 16);   // K0+16..K0+23
      acc = WMMA_BF16(af.v, bfr.v, acc);
    }
  }

  const int n = ln;                 // D column = group-local oc
  if (n < MAXO_) {
    float bias = b2[g * MAXO_ + n];
#pragma unroll
    for (int r = 0; r < 8; r++) {
      int m = r + hi * 8;           // D row = pixel in tile
      int pp = pixBase + m;
      int bb = pp / HW, rr = pp % HW;
      y2[((size_t)bb * OC2 + (g * MAXO_ + n)) * HW + rr] = acc[r] + bias;
    }
  }
}

// ---------------- heads / maps ----------------
__global__ void maps_kernel(const float* __restrict__ y2, float* __restrict__ dout) {
  int t = blockIdx.x * blockDim.x + threadIdx.x;
  if (t >= NPIX) return;
  int s = t / HW, r = t % HW;
  const float* yb = y2 + (size_t)s * OC2 * HW;
  dout[O_LMAP + t] = sigm(yb[0 * HW + r]);
  dout[O_JMAP + t] = sigm(yb[4 * HW + r]);
  dout[O_JOFF + s * 2 * HW + r]          = yb[8 * HW + r];
  dout[O_JOFF + s * 2 * HW + HW + r]     = yb[9 * HW + r];
  dout[O_CMAP + t] = sigm(yb[12 * HW + r]);
  dout[O_COFF + s * 2 * HW + r]          = yb[16 * HW + r];
  dout[O_COFF + s * 2 * HW + HW + r]     = yb[17 * HW + r];
  dout[O_LVEC + s * 4 * HW + 0 * HW + r] = yb[20 * HW + r];
  dout[O_LVEC + s * 4 * HW + 1 * HW + r] = yb[21 * HW + r];
  dout[O_LVEC + s * 4 * HW + 2 * HW + r] = yb[22 * HW + r];
  dout[O_LVEC + s * 4 * HW + 3 * HW + r] = yb[23 * HW + r];
}

// ---------------- NMS ----------------
__global__ void nms_kernel(const float* __restrict__ jmap, float* __restrict__ out) {
  int t = blockIdx.x * blockDim.x + threadIdx.x;
  if (t >= NPIX) return;
  int s = t / HW, r = t % HW, y = r / W_, x = r % W_;
  const float* p = jmap + (size_t)s * HW;
  float v = p[r], mx = -3.402823466e38f;
  for (int dy = -1; dy <= 1; dy++)
    for (int dx = -1; dx <= 1; dx++) {
      int yy = y + dy, xx = x + dx;
      if (yy >= 0 && yy < H_ && xx >= 0 && xx < W_) mx = fmaxf(mx, p[yy * W_ + xx]);
    }
  out[t] = (v == mx) ? v : 0.0f;
}

// ---------------- top-k via bitonic sort (32768 entries in LDS) ----------------
__global__ void topk_kernel(const float* __restrict__ src, float* __restrict__ outV,
                            int* __restrict__ outI, int K) {
  extern __shared__ char tks[];
  float* sv = (float*)tks;
  int* si = (int*)(tks + 32768 * sizeof(float));
  int s = blockIdx.x;
  const float* p = src + (size_t)s * HW;
  for (int i = threadIdx.x; i < 32768; i += blockDim.x) {
    sv[i] = (i < HW) ? p[i] : -3.402823466e38f;
    si[i] = i;
  }
  __syncthreads();
  for (int k = 2; k <= 32768; k <<= 1) {
    for (int j = k >> 1; j > 0; j >>= 1) {
      for (int i = threadIdx.x; i < 32768; i += blockDim.x) {
        int ixj = i ^ j;
        if (ixj > i) {
          bool up = ((i & k) == 0);
          float a = sv[i], b = sv[ixj];
          bool sw = up ? (a < b) : (a > b);   // descending overall
          if (sw) {
            sv[i] = b; sv[ixj] = a;
            int t = si[i]; si[i] = si[ixj]; si[ixj] = t;
          }
        }
      }
      __syncthreads();
    }
  }
  for (int i = threadIdx.x; i < K; i += blockDim.x) {
    outV[s * K + i] = sv[i];
    outI[s * K + i] = si[i];
  }
}

// ---------------- junction coords ----------------
__global__ void junc_kernel(const float* __restrict__ jVals, const int* __restrict__ jIdx,
                            const float* __restrict__ dout,
                            float* __restrict__ jx, float* __restrict__ jy,
                            int* __restrict__ jvalid) {
  int s = blockIdx.x, i = threadIdx.x;
  if (i >= JK) return;
  float v = jVals[s * JK + i];
  int idx = jIdx[s * JK + i];
  const float* joff = dout + O_JOFF + s * 2 * HW;
  float X = (float)(idx % W_) + joff[idx] + 0.5f;
  float Y = (float)(idx / W_) + joff[HW + idx] + 0.5f;
  X = fminf(fmaxf(X, 0.0f), (float)W_ - 1e-4f);
  Y = fminf(fmaxf(Y, 0.0f), (float)H_ - 1e-4f);
  jx[s * JK + i] = X;
  jy[s * JK + i] = Y;
  jvalid[s * JK + i] = (v >= 0.008f) ? 1 : 0;
}

// ---------------- line endpoint / junction matching ----------------
__global__ void match_kernel(const int* __restrict__ cIdx, const float* __restrict__ dout,
                             const float* __restrict__ jx, const float* __restrict__ jy,
                             const int* __restrict__ jvalid, int* __restrict__ keys) {
  __shared__ float sjx[JK], sjy[JK];
  __shared__ int sjv[JK];
  int s = blockIdx.y;
  for (int i = threadIdx.x; i < JK; i += blockDim.x) {
    sjx[i] = jx[s * JK + i];
    sjy[i] = jy[s * JK + i];
    sjv[i] = jvalid[s * JK + i];
  }
  __syncthreads();
  int line = blockIdx.x * blockDim.x + threadIdx.x;
  if (line >= LK) return;
  int idx = cIdx[s * LK + line];
  const float* coff = dout + O_COFF + s * 2 * HW;
  const float* lvec = dout + O_LVEC + s * 4 * HW;
  float cx = (float)(idx % W_) + coff[idx] + 0.5f;
  float cy = (float)(idx / W_) + coff[HW + idx] + 0.5f;
  float x0 = cx + lvec[idx],          y0 = cy + lvec[HW + idx];
  float x1 = cx + lvec[2 * HW + idx], y1 = cy + lvec[3 * HW + idx];
  x0 = fminf(fmaxf(x0, 0.0f), (float)W_ - 1e-4f);
  y0 = fminf(fmaxf(y0, 0.0f), (float)H_ - 1e-4f);
  x1 = fminf(fmaxf(x1, 0.0f), (float)W_ - 1e-4f);
  y1 = fminf(fmaxf(y1, 0.0f), (float)H_ - 1e-4f);
  float bd1 = 1e18f, bd2 = 1e18f;
  int i1 = 0, i2 = 0;
  for (int j = 0; j < JK; j++) {
    float dx0 = x0 - sjx[j], dy0 = y0 - sjy[j];
    float dx1 = x1 - sjx[j], dy1 = y1 - sjy[j];
    float d1 = dx0 * dx0 + dy0 * dy0;
    float d2 = dx1 * dx1 + dy1 * dy1;
    if (!sjv[j]) { d1 = 1e18f; d2 = 1e18f; }
    if (d1 < bd1) { bd1 = d1; i1 = j; }
    if (d2 < bd2) { bd2 = d2; i2 = j; }
  }
  int lo = min(i1, i2), hi = max(i1, i2);
  keys[s * 8192 + line] = (lo != hi) ? (lo * JK + hi) : BIGK;
}

// ---------------- sort keys ascending (bitonic, 8192) ----------------
__global__ void sortkeys_kernel(int* __restrict__ keys) {
  __shared__ int sk[8192];
  int s = blockIdx.x;
  int* p = keys + s * 8192;
  for (int i = threadIdx.x; i < 8192; i += blockDim.x)
    sk[i] = (i < LK) ? p[i] : BIGK;
  __syncthreads();
  for (int k = 2; k <= 8192; k <<= 1) {
    for (int j = k >> 1; j > 0; j >>= 1) {
      for (int i = threadIdx.x; i < 8192; i += blockDim.x) {
        int ixj = i ^ j;
        if (ixj > i) {
          bool up = ((i & k) == 0);
          int a = sk[i], b = sk[ixj];
          if (up ? (a > b) : (a < b)) { sk[i] = b; sk[ixj] = a; }
        }
      }
      __syncthreads();
    }
  }
  for (int i = threadIdx.x; i < 8192; i += blockDim.x) p[i] = sk[i];
}

// ---------------- finalize: dedupe + emit loi/mask ----------------
__global__ void finalize_kernel(const int* __restrict__ keys,
                                const float* __restrict__ jx, const float* __restrict__ jy,
                                float* __restrict__ dout) {
  int s = blockIdx.x;
  const int* sk = keys + s * 8192;
  for (int i = threadIdx.x; i < LK; i += blockDim.x) {
    int kc = sk[i];
    int kp = (i > 0) ? sk[i - 1] : (BIGK + 1);
    bool first = ((i == 0) || (kc != kp)) && (kc < BIGK);
    int a = first ? kc / JK : 0;
    int b = first ? kc % JK : 0;
    float ax = jx[s * JK + a], ay = jy[s * JK + a];
    float bx = jx[s * JK + b], by = jy[s * JK + b];
    bool sw = ay > by;
    float* loi = dout + O_LOI + (size_t)((s * LK + i) * 2) * 2;
    loi[0] = sw ? bx : ax;
    loi[1] = sw ? by : ay;
    loi[2] = sw ? ax : bx;
    loi[3] = sw ? ay : by;
    dout[O_MASK + s * LK + i] = first ? 1.0f : 0.0f;
  }
}

// ---------------- workspace layout (bytes) ----------------
#define WS_FEAT   ((size_t)0)                            // B_*PHW_*C_*2  = 53,747,712
#define WS_ACT    (WS_FEAT + (size_t)B_ * PHW_ * C_ * 2) // B_*PHW_*OC1*2 = 80,621,568
#define WS_PW1    (WS_ACT + (size_t)B_ * PHW_ * OC1 * 2)
#define WS_BIAS1  (WS_PW1 + (size_t)24 * K1C * 512 * 2)
#define WS_PW2    (WS_BIAS1 + (size_t)OC1 * 4)
#define WS_Y2     (WS_PW2 + (size_t)NH_ * K2C * 512 * 2)
#define WS_NMS    (WS_Y2 + (size_t)B_ * OC2 * HW * 4)
#define WS_JVALS  (WS_NMS + (size_t)NPIX * 4)
#define WS_JIDX   (WS_JVALS + (size_t)B_ * JK * 4)
#define WS_JX     (WS_JIDX + (size_t)B_ * JK * 4)
#define WS_JY     (WS_JX + (size_t)B_ * JK * 4)
#define WS_JVALID (WS_JY + (size_t)B_ * JK * 4)
#define WS_CVALS  (WS_JVALID + (size_t)B_ * JK * 4)
#define WS_CIDX   (WS_CVALS + (size_t)B_ * LK * 4)
#define WS_KEYS   (WS_CIDX + (size_t)B_ * LK * 4)

extern "C" void kernel_launch(void* const* d_in, const int* in_sizes, int n_in,
                              void* d_out, int out_size, void* d_ws, size_t ws_size,
                              hipStream_t stream) {
  const float* features = (const float*)d_in[0];
  const float* w1    = (const float*)d_in[1];
  const float* b1    = (const float*)d_in[2];
  const float* gamma = (const float*)d_in[3];
  const float* beta  = (const float*)d_in[4];
  const float* mean  = (const float*)d_in[5];
  const float* var   = (const float*)d_in[6];
  const float* w2    = (const float*)d_in[7];
  const float* b2    = (const float*)d_in[8];
  float* dout = (float*)d_out;
  char* ws = (char*)d_ws;

  unsigned short* featP  = (unsigned short*)(ws + WS_FEAT);
  unsigned short* actP   = (unsigned short*)(ws + WS_ACT);
  unsigned short* pw1    = (unsigned short*)(ws + WS_PW1);
  float*          bias1  = (float*)(ws + WS_BIAS1);
  unsigned short* pw2    = (unsigned short*)(ws + WS_PW2);
  float*          y2     = (float*)(ws + WS_Y2);
  float*          nmsb   = (float*)(ws + WS_NMS);
  float*          jVals  = (float*)(ws + WS_JVALS);
  int*            jIdx   = (int*)(ws + WS_JIDX);
  float*          jx     = (float*)(ws + WS_JX);
  float*          jy     = (float*)(ws + WS_JY);
  int*            jvalid = (int*)(ws + WS_JVALID);
  float*          cVals  = (float*)(ws + WS_CVALS);
  int*            cIdx   = (int*)(ws + WS_CIDX);
  int*            keys   = (int*)(ws + WS_KEYS);

  // weight / activation packing
  bias_fold_kernel<<<(OC1 + 255) / 256, 256, 0, stream>>>(b1, gamma, beta, mean, var, bias1);
  w1_pack_kernel<<<(24 * K1C * 512 + 255) / 256, 256, 0, stream>>>(w1, gamma, var, pw1);
  w2_pack_kernel<<<(NH_ * K2C * 512 + 255) / 256, 256, 0, stream>>>(w2, pw2);
  feat_pack_kernel<<<(B_ * PHW_ * C_ + 255) / 256, 256, 0, stream>>>(features, featP);
  {
    size_t q = (size_t)B_ * PHW_ * OC1 / 8;
    act_zero_kernel<<<(unsigned)((q + 255) / 256), 256, 0, stream>>>((uint4*)actP);
  }

  // convs (WMMA)
  conv1_wmma_kernel<<<dim3(800, 3), 256, 0, stream>>>(featP, pw1, bias1, actP);
  conv2_wmma_kernel<<<dim3(800, NH_), 256, 0, stream>>>(actP, pw2, b2, y2);

  // heads
  maps_kernel<<<(NPIX + 255) / 256, 256, 0, stream>>>(y2, dout);

  // post-processing
  nms_kernel<<<(NPIX + 255) / 256, 256, 0, stream>>>(dout + O_JMAP, nmsb);
  topk_kernel<<<B_, 512, 32768 * 8, stream>>>(nmsb, jVals, jIdx, JK);
  junc_kernel<<<B_, 320, 0, stream>>>(jVals, jIdx, dout, jx, jy, jvalid);
  topk_kernel<<<B_, 512, 32768 * 8, stream>>>(dout + O_CMAP, cVals, cIdx, LK);
  match_kernel<<<dim3((LK + 255) / 256, B_), 256, 0, stream>>>(cIdx, dout, jx, jy, jvalid, keys);
  sortkeys_kernel<<<B_, 256, 0, stream>>>(keys);
  finalize_kernel<<<B_, 256, 0, stream>>>(keys, jx, jy, dout);
}
